// VSSBlock_1D_27943057228198
// MI455X (gfx1250) — matrix operand order repaired and checked by reading, not compiled, not run on hardware
//
#include <hip/hip_runtime.h>
#include <hip/hip_bf16.h>

typedef __attribute__((ext_vector_type(16))) _Float16 v16h;
typedef __attribute__((ext_vector_type(8)))  _Float16 v8h;
typedef __attribute__((ext_vector_type(8)))  float    v8f;

#define MTOK 16384   // B*H*W = 4*64*64
#define NCHUNK 64
#define CLEN 64      // 4096 / 64

__device__ __forceinline__ float siluf(float v)     { return v / (1.f + __expf(-v)); }
__device__ __forceinline__ float softplusf(float v) { return (v > 20.f) ? v : log1pf(__expf(v)); }

// ---------------------------------------------------------------------------
// Generic tiled WMMA GEMM:  C[M,N] = A[M,K](f16,row-major) * W[N,K]^T (f16)
// Block = 256 threads = 8 waves; wave tile 16x16; block tile 32x64.
// Epilogues: 0=f32 store, 1=f16 store, 2=proj(main+bias | gate silu+bias),
//            3=softplus(dt)+bias -> f32, 4=out_proj bias+residual -> NCHW.
// ---------------------------------------------------------------------------
struct EpiArgs {
  float*          out32;
  _Float16*       out16a;
  _Float16*       out16b;
  const float*    bias;
  const float*    resid;
  int             ldo;
};

template<int EPI>
__global__ __launch_bounds__(256)
void wmma_gemm(const _Float16* __restrict__ A, int lda,
               const _Float16* __restrict__ Bw, int ldb,
               int kiters, EpiArgs e)
{
  const int lane = threadIdx.x & 31;
  const int wave = threadIdx.x >> 5;
  const int m0 = blockIdx.y * 32 + (wave & 1) * 16;
  const int n0 = blockIdx.x * 64 + (wave >> 1) * 16;
  const int h  = lane >> 4;     // lane half
  const int r  = lane & 15;

  // A fragment (16x32 f16): lane half 0 holds K 0..7 & 16..23, half 1 holds 8..15 & 24..31
  const _Float16* ap = A  + (size_t)(m0 + r) * lda + h * 8;
  // B fragment (32x16 f16): column N = r, lane half 0 holds K 0..7 & 16..23, half 1 holds 8..15 & 24..31
  const _Float16* bp = Bw + (size_t)(n0 + r) * ldb + h * 8;

  v8f c = {};
  for (int k = 0; k < kiters; ++k) {
    v8h a0 = *(const v8h*)(ap);
    v8h a1 = *(const v8h*)(ap + 16);
    v16h av = __builtin_shufflevector(a0, a1, 0,1,2,3,4,5,6,7,8,9,10,11,12,13,14,15);
    v8h b0 = *(const v8h*)(bp);
    v8h b1 = *(const v8h*)(bp + 16);
    v16h bv = __builtin_shufflevector(b0, b1, 0,1,2,3,4,5,6,7,8,9,10,11,12,13,14,15);
    __builtin_prefetch(ap + 32, 0, 0);   // global_prefetch_b8
    c = __builtin_amdgcn_wmma_f32_16x16x32_f16(false, av, false, bv, (short)0, c, false, false);
    ap += 32; bp += 32;
  }

  // C/D layout: VGPR i -> M = m0 + 8*h + i ; N = n0 + r
  #pragma unroll
  for (int i = 0; i < 8; ++i) {
    const int m = m0 + h * 8 + i;
    const int n = n0 + r;
    float v = c[i];
    if constexpr (EPI == 0) {
      e.out32[(size_t)m * e.ldo + n] = v;
    } else if constexpr (EPI == 1) {
      e.out16a[(size_t)m * e.ldo + n] = (_Float16)v;
    } else if constexpr (EPI == 2) {
      v += e.bias[n];
      if (n < 384) e.out16a[(size_t)m * 384 + n]         = (_Float16)v;
      else         e.out16b[(size_t)m * 384 + (n - 384)] = (_Float16)siluf(v);
    } else if constexpr (EPI == 3) {
      e.out32[(size_t)m * 384 + n] = softplusf(v + e.bias[n]);
    } else if constexpr (EPI == 4) {
      const int b = m >> 12, hw = m & 4095;
      const size_t o = ((size_t)(b * 192 + n) << 12) + hw;
      e.out32[o] = v + e.bias[n] + e.resid[o];
    }
  }
}

// ---------------------------------------------------------------------------
// Weight prep: f32 -> f16 with row/col zero padding
// ---------------------------------------------------------------------------
__global__ void k_cvt_f16(const float* __restrict__ src, _Float16* __restrict__ dst,
                          int rdst, int rsrc, int csrc, int cdst)
{
  int i = blockIdx.x * 256 + threadIdx.x;
  if (i >= rdst * cdst) return;
  int r = i / cdst, c = i - r * cdst;
  float v = (r < rsrc && c < csrc) ? src[(size_t)r * csrc + c] : 0.f;
  dst[i] = (_Float16)v;
}

__global__ void k_concat2(const float* a, const float* b, float* dst)
{
  int i = blockIdx.x * 256 + threadIdx.x;
  if (i < 384) dst[i] = a[i];
  else if (i < 768) dst[i] = b[i - 384];
}

__global__ void k_aneg(const float* alog, float* aneg)
{
  int i = blockIdx.x * 256 + threadIdx.x;
  if (i < 384 * 16) aneg[i] = -__expf(alog[i]);
}

// ---------------------------------------------------------------------------
// AdaLN conditioning (tiny: B=4)
// ---------------------------------------------------------------------------
__global__ void k_adaln1(const float* cond, const float* w1, const float* b1, float* hid)
{
  int i = blockIdx.x * 256 + threadIdx.x;
  if (i >= 4 * 192) return;
  int b = i / 192, c = i - b * 192;
  float acc = b1[c];
  for (int k = 0; k < 256; ++k) acc += cond[b * 256 + k] * w1[c * 256 + k];
  hid[i] = siluf(acc);
}

__global__ void k_adaln2(const float* hid, const float* w2, const float* b2, float* gb)
{
  int i = blockIdx.x * 256 + threadIdx.x;
  if (i >= 4 * 384) return;
  int b = i / 384, j = i - b * 384;
  float acc = b2[j];
  for (int k = 0; k < 192; ++k) acc += hid[b * 192 + k] * w2[j * 192 + k];
  gb[i] = acc;
}

// ---------------------------------------------------------------------------
// LayerNorm(C=192) + AdaLN modulation; one wave per token, shuffle reduce.
// x is NCHW; output row-major (token, 192) f16.
// ---------------------------------------------------------------------------
__global__ __launch_bounds__(256)
void k_norm_mod(const float* __restrict__ x, const float* __restrict__ nw,
                const float* __restrict__ nb, const float* __restrict__ gb,
                _Float16* __restrict__ xn)
{
  int wid  = (blockIdx.x * 256 + threadIdx.x) >> 5;
  int lane = threadIdx.x & 31;
  if (wid >= MTOK) return;
  int b = wid >> 12, hw = wid & 4095;
  float vals[6], s = 0.f, s2 = 0.f;
  #pragma unroll
  for (int j = 0; j < 6; ++j) {
    int c = lane + j * 32;
    float v = x[((size_t)(b * 192 + c) << 12) + hw];
    vals[j] = v; s += v; s2 += v * v;
  }
  for (int o = 16; o; o >>= 1) { s += __shfl_xor(s, o, 32); s2 += __shfl_xor(s2, o, 32); }
  float mu = s * (1.f / 192.f);
  float rs = rsqrtf(s2 * (1.f / 192.f) - mu * mu + 1e-5f);
  #pragma unroll
  for (int j = 0; j < 6; ++j) {
    int c = lane + j * 32;
    float v = (vals[j] - mu) * rs * nw[c] + nb[c];
    v = v * (1.f + gb[b * 384 + c]) + gb[b * 384 + 192 + c];
    xn[(size_t)wid * 192 + c] = (_Float16)v;
  }
}

// ---------------------------------------------------------------------------
// Depthwise 3x3 conv (SAME) + bias + SiLU, NHWC-flat f16 -> f16
// ---------------------------------------------------------------------------
__global__ __launch_bounds__(256)
void k_dwconv(const _Float16* __restrict__ src, const float* __restrict__ kw,
              const float* __restrict__ kb, _Float16* __restrict__ dst)
{
  int i = blockIdx.x * 256 + threadIdx.x;
  if (i >= MTOK * 384) return;
  int m = i / 384, c = i - m * 384;
  int b = m >> 12, hw = m & 4095, hh = hw >> 6, ww = hw & 63;
  float acc = kb[c];
  #pragma unroll
  for (int dy = 0; dy < 3; ++dy) {
    int yy = hh + dy - 1;
    if (yy < 0 || yy > 63) continue;
    #pragma unroll
    for (int dx = 0; dx < 3; ++dx) {
      int xx = ww + dx - 1;
      if (xx < 0 || xx > 63) continue;
      acc += (float)src[((size_t)((b << 12) + (yy << 6) + xx)) * 384 + c] * kw[c * 9 + dy * 3 + dx];
    }
  }
  dst[(size_t)m * 384 + c] = (_Float16)siluf(acc);
}

// ---------------------------------------------------------------------------
// Causal depthwise conv1d (k=4) over L + bias + SiLU.  u = xz[:, :384]
// ---------------------------------------------------------------------------
__global__ __launch_bounds__(256)
void k_conv1d(const _Float16* __restrict__ xz, const float* __restrict__ kw,
              const float* __restrict__ kb, _Float16* __restrict__ uc)
{
  int i = blockIdx.x * 256 + threadIdx.x;
  if (i >= MTOK * 384) return;
  int m = i / 384, c = i - m * 384;
  int b = m >> 12, t = m & 4095;
  float acc = kb[c];
  #pragma unroll
  for (int j = 0; j < 4; ++j) {
    int tt = t - 3 + j;
    if (tt >= 0) acc += (float)xz[((size_t)((b << 12) + tt)) * 768 + c] * kw[c * 4 + j];
  }
  uc[(size_t)m * 384 + c] = (_Float16)siluf(acc);
}

// ---------------------------------------------------------------------------
// Chunked parallel selective scan.
// Pass1: per (batch,chunk) block (384 threads, one per d): compute
//        P[n] = prod dA, q[n] = h after chunk with h0=0. B/C staged in LDS.
// ---------------------------------------------------------------------------
__global__ __launch_bounds__(384)
void k_scan1(const float* __restrict__ dt, const _Float16* __restrict__ uc,
             const _Float16* __restrict__ dbl, const float* __restrict__ aneg,
             float* __restrict__ pq)
{
  __shared__ float sBC[CLEN * 32];
  const int blk = blockIdx.x;             // b*NCHUNK + chunk
  const int b = blk >> 6, ch = blk & 63;
  const int m0 = (b << 12) + (ch << 6);
  for (int idx = threadIdx.x; idx < CLEN * 32; idx += 384) {
    int t = idx >> 5, j = idx & 31;       // cols 24..55 of dbl = [B(16) | C(16)]
    sBC[idx] = (float)dbl[(size_t)(m0 + t) * 64 + 24 + j];
  }
  __syncthreads();
  const int d = threadIdx.x;
  float a[16], h[16], P[16];
  #pragma unroll
  for (int n = 0; n < 16; ++n) { a[n] = aneg[d * 16 + n]; h[n] = 0.f; P[n] = 1.f; }
  for (int t = 0; t < CLEN; ++t) {
    const int m = m0 + t;
    const float dtv = dt[(size_t)m * 384 + d];
    const float du  = dtv * (float)uc[(size_t)m * 384 + d];
    const float* bc = &sBC[t * 32];
    #pragma unroll
    for (int n = 0; n < 16; ++n) {
      float dA = __expf(dtv * a[n]);
      h[n] = dA * h[n] + du * bc[n];
      P[n] *= dA;
    }
  }
  const size_t base = ((size_t)blk * 384 + d) * 32;
  #pragma unroll
  for (int n = 0; n < 16; ++n) { pq[base + n] = P[n]; pq[base + 16 + n] = h[n]; }
}

// Pass2: chunk-level combine. Thread per (b,d,n), 64 sequential chunk steps.
__global__ __launch_bounds__(256)
void k_scan2(const float* __restrict__ pq, float* __restrict__ h0)
{
  int i = blockIdx.x * 256 + threadIdx.x;
  if (i >= 4 * 384 * 16) return;
  int n = i & 15, d = (i >> 4) % 384, b = i / (384 * 16);
  float h = 0.f;
  for (int c = 0; c < NCHUNK; ++c) {
    const size_t idx = ((size_t)((b << 6) + c) * 384 + d);
    h0[(idx << 4) + n] = h;
    h = pq[(idx << 5) + n] * h + pq[(idx << 5) + 16 + n];
  }
}

// Pass3: replay with correct initial state, emit y = h . C
__global__ __launch_bounds__(384)
void k_scan3(const float* __restrict__ dt, const _Float16* __restrict__ uc,
             const _Float16* __restrict__ dbl, const float* __restrict__ aneg,
             const float* __restrict__ h0, float* __restrict__ y)
{
  __shared__ float sBC[CLEN * 32];
  const int blk = blockIdx.x;
  const int b = blk >> 6, ch = blk & 63;
  const int m0 = (b << 12) + (ch << 6);
  for (int idx = threadIdx.x; idx < CLEN * 32; idx += 384) {
    int t = idx >> 5, j = idx & 31;
    sBC[idx] = (float)dbl[(size_t)(m0 + t) * 64 + 24 + j];
  }
  __syncthreads();
  const int d = threadIdx.x;
  float a[16], h[16];
  #pragma unroll
  for (int n = 0; n < 16; ++n) {
    a[n] = aneg[d * 16 + n];
    h[n] = h0[((size_t)blk * 384 + d) * 16 + n];
  }
  for (int t = 0; t < CLEN; ++t) {
    const int m = m0 + t;
    const float dtv = dt[(size_t)m * 384 + d];
    const float du  = dtv * (float)uc[(size_t)m * 384 + d];
    const float* bc = &sBC[t * 32];
    float acc = 0.f;
    #pragma unroll
    for (int n = 0; n < 16; ++n) {
      float dA = __expf(dtv * a[n]);
      h[n] = dA * h[n] + du * bc[n];
      acc += h[n] * bc[16 + n];
    }
    y[(size_t)m * 384 + d] = acc;
  }
}

// y = (y_scan + u*D) * silu(z)  -> f16 for mamba_out GEMM
__global__ __launch_bounds__(256)
void k_ymix(const float* __restrict__ y32, const _Float16* __restrict__ uc,
            const float* __restrict__ Dv, const _Float16* __restrict__ xz,
            _Float16* __restrict__ y16)
{
  int i = blockIdx.x * 256 + threadIdx.x;
  if (i >= MTOK * 384) return;
  int m = i / 384, c = i - m * 384;
  float y = y32[(size_t)m * 384 + c] + (float)uc[(size_t)m * 384 + c] * Dv[c];
  float z = (float)xz[(size_t)m * 768 + 384 + c];
  y16[(size_t)m * 384 + c] = (_Float16)(y * siluf(z));
}

// LayerNorm(384) over mamba output, then * gate -> f16 for out_proj GEMM
__global__ __launch_bounds__(256)
void k_lngate(const float* __restrict__ mo, const float* __restrict__ nw,
              const float* __restrict__ nb, const _Float16* __restrict__ gate,
              _Float16* __restrict__ ln)
{
  int wid  = (blockIdx.x * 256 + threadIdx.x) >> 5;
  int lane = threadIdx.x & 31;
  if (wid >= MTOK) return;
  float vals[12], s = 0.f, s2 = 0.f;
  #pragma unroll
  for (int j = 0; j < 12; ++j) {
    int c = lane + j * 32;
    float v = mo[(size_t)wid * 384 + c];
    vals[j] = v; s += v; s2 += v * v;
  }
  for (int o = 16; o; o >>= 1) { s += __shfl_xor(s, o, 32); s2 += __shfl_xor(s2, o, 32); }
  float mu = s * (1.f / 384.f);
  float rs = rsqrtf(s2 * (1.f / 384.f) - mu * mu + 1e-5f);
  #pragma unroll
  for (int j = 0; j < 12; ++j) {
    int c = lane + j * 32;
    float v = (vals[j] - mu) * rs * nw[c] + nb[c];
    ln[(size_t)wid * 384 + c] = (_Float16)(v * (float)gate[(size_t)wid * 384 + c]);
  }
}

// ---------------------------------------------------------------------------
extern "C" void kernel_launch(void* const* d_in, const int* in_sizes, int n_in,
                              void* d_out, int out_size, void* d_ws, size_t ws_size,
                              hipStream_t stream)
{
  (void)in_sizes; (void)n_in; (void)out_size; (void)ws_size;
  const float* x    = (const float*)d_in[0];
  const float* cond = (const float*)d_in[1];
  const float* nw   = (const float*)d_in[2];
  const float* nb   = (const float*)d_in[3];
  const float* aw1  = (const float*)d_in[4];
  const float* ab1  = (const float*)d_in[5];
  const float* aw2  = (const float*)d_in[6];
  const float* ab2  = (const float*)d_in[7];
  const float* pmw  = (const float*)d_in[8];
  const float* pmb  = (const float*)d_in[9];
  const float* pgw  = (const float*)d_in[10];
  const float* pgb  = (const float*)d_in[11];
  const float* dww  = (const float*)d_in[12];
  const float* dwb  = (const float*)d_in[13];
  const float* ipw  = (const float*)d_in[14];
  const float* c1w  = (const float*)d_in[15];
  const float* c1b  = (const float*)d_in[16];
  const float* xpw  = (const float*)d_in[17];
  const float* dtw  = (const float*)d_in[18];
  const float* dtb  = (const float*)d_in[19];
  const float* alog = (const float*)d_in[20];
  const float* Dv   = (const float*)d_in[21];
  const float* mow  = (const float*)d_in[22];
  const float* nmw  = (const float*)d_in[23];
  const float* nmb  = (const float*)d_in[24];
  const float* opw  = (const float*)d_in[25];
  const float* opb  = (const float*)d_in[26];
  float* out = (float*)d_out;

  char* w = (char*)d_ws;
  size_t off = 0;
  auto alloc = [&](size_t bytes) { size_t o = off; off += (bytes + 255) & ~(size_t)255; return o; };

  _Float16* WC16  = (_Float16*)(w + alloc((size_t)768 * 192 * 2));
  float*    BIASC = (float*)   (w + alloc(768 * 4));
  _Float16* W_INP = (_Float16*)(w + alloc((size_t)768 * 384 * 2));
  _Float16* W_XP  = (_Float16*)(w + alloc((size_t)64  * 384 * 2));
  _Float16* W_DT  = (_Float16*)(w + alloc((size_t)384 * 32  * 2));
  _Float16* W_MO  = (_Float16*)(w + alloc((size_t)384 * 384 * 2));
  _Float16* W_OP  = (_Float16*)(w + alloc((size_t)192 * 384 * 2));
  float*    ANEG  = (float*)   (w + alloc(384 * 16 * 4));
  float*    HIDB  = (float*)   (w + alloc(4 * 192 * 4));
  float*    GB    = (float*)   (w + alloc(4 * 384 * 4));
  // big regions with stage-aliasing (lifetimes are disjoint)
  char* R_XN   = w + alloc((size_t)MTOK * 192 * 2);   // xn16  -> dbl16
  char* R_MAIN = w + alloc((size_t)MTOK * 384 * 2);   // main16 -> uc16
  char* R_GATE = w + alloc((size_t)MTOK * 384 * 2);   // gate16 (live to end)
  char* R_SEQ  = w + alloc((size_t)MTOK * 384 * 2);   // seq16 -> PQ -> y16
  char* R_XZ   = w + alloc((size_t)MTOK * 768 * 2);   // xz16
  char* R_DT   = w + alloc((size_t)MTOK * 384 * 4);   // dt32 -> mo32
  char* R_H0   = w + alloc((size_t)256 * 384 * 16 * 4);
  char* R_Y    = w + alloc((size_t)MTOK * 384 * 4);   // y32 -> ln16

  _Float16* XN16   = (_Float16*)R_XN;
  _Float16* DBL16  = (_Float16*)R_XN;
  _Float16* MAIN16 = (_Float16*)R_MAIN;
  _Float16* UC16   = (_Float16*)R_MAIN;
  _Float16* GATE16 = (_Float16*)R_GATE;
  _Float16* SEQ16  = (_Float16*)R_SEQ;
  float*    PQ     = (float*)R_SEQ;
  _Float16* Y16    = (_Float16*)R_SEQ;
  _Float16* XZ16   = (_Float16*)R_XZ;
  float*    DT32   = (float*)R_DT;
  float*    MO32   = (float*)R_DT;
  float*    H0     = (float*)R_H0;
  float*    Y32    = (float*)R_Y;
  _Float16* LN16   = (_Float16*)R_Y;

  auto g1 = [](int n) { return dim3((unsigned)((n + 255) / 256)); };

  // --- weight prep / conditioning ---
  k_cvt_f16<<<g1(384 * 192), 256, 0, stream>>>(pmw, WC16,             384, 384, 192, 192);
  k_cvt_f16<<<g1(384 * 192), 256, 0, stream>>>(pgw, WC16 + 384 * 192, 384, 384, 192, 192);
  k_concat2<<<3, 256, 0, stream>>>(pmb, pgb, BIASC);
  k_cvt_f16<<<g1(768 * 384), 256, 0, stream>>>(ipw, W_INP, 768, 768, 384, 384);
  k_cvt_f16<<<g1(64  * 384), 256, 0, stream>>>(xpw, W_XP,  64,  56,  384, 384);
  k_cvt_f16<<<g1(384 * 32 ), 256, 0, stream>>>(dtw, W_DT,  384, 384, 24,  32);
  k_cvt_f16<<<g1(384 * 384), 256, 0, stream>>>(mow, W_MO,  384, 384, 384, 384);
  k_cvt_f16<<<g1(192 * 384), 256, 0, stream>>>(opw, W_OP,  192, 192, 384, 384);
  k_aneg<<<g1(384 * 16), 256, 0, stream>>>(alog, ANEG);
  k_adaln1<<<g1(4 * 192), 256, 0, stream>>>(cond, aw1, ab1, HIDB);
  k_adaln2<<<g1(4 * 384), 256, 0, stream>>>(HIDB, aw2, ab2, GB);

  // --- pipeline ---
  k_norm_mod<<<2048, 256, 0, stream>>>(x, nw, nb, GB, XN16);
  { EpiArgs e{}; e.out16a = MAIN16; e.out16b = GATE16; e.bias = BIASC;
    wmma_gemm<2><<<dim3(12, 512), 256, 0, stream>>>(XN16, 192, WC16, 192, 6, e); }
  k_dwconv<<<g1(MTOK * 384), 256, 0, stream>>>(MAIN16, dww, dwb, SEQ16);
  { EpiArgs e{}; e.out16a = XZ16; e.ldo = 768;
    wmma_gemm<1><<<dim3(12, 512), 256, 0, stream>>>(SEQ16, 384, W_INP, 384, 12, e); }
  k_conv1d<<<g1(MTOK * 384), 256, 0, stream>>>(XZ16, c1w, c1b, UC16);
  { EpiArgs e{}; e.out16a = DBL16; e.ldo = 64;
    wmma_gemm<1><<<dim3(1, 512), 256, 0, stream>>>(UC16, 384, W_XP, 384, 12, e); }
  { EpiArgs e{}; e.out32 = DT32; e.bias = dtb;
    wmma_gemm<3><<<dim3(6, 512), 256, 0, stream>>>(DBL16, 64, W_DT, 32, 1, e); }
  k_scan1<<<256, 384, 0, stream>>>(DT32, UC16, DBL16, ANEG, PQ);
  k_scan2<<<g1(4 * 384 * 16), 256, 0, stream>>>(PQ, H0);
  k_scan3<<<256, 384, 0, stream>>>(DT32, UC16, DBL16, ANEG, H0, Y32);
  k_ymix<<<g1(MTOK * 384), 256, 0, stream>>>(Y32, UC16, Dv, XZ16, Y16);
  { EpiArgs e{}; e.out32 = MO32; e.ldo = 384;
    wmma_gemm<0><<<dim3(6, 512), 256, 0, stream>>>(Y16, 384, W_MO, 384, 12, e); }
  k_lngate<<<2048, 256, 0, stream>>>(MO32, nmw, nmb, GATE16, LN16);
  { EpiArgs e{}; e.out32 = out; e.bias = opb; e.resid = x;
    wmma_gemm<4><<<dim3(3, 512), 256, 0, stream>>>(LN16, 384, W_OP, 384, 12, e); }
}
